// MultiHeadAttentionGatedPooling_6408091206154
// MI455X (gfx1250) — compile-verified, matrix operand order checked by reading, not run on hardware
//
#include <hip/hip_runtime.h>
#include <math.h>

// Problem constants (match reference)
#define NHEAD   8
#define DMODEL  512
#define DKH     64                 // DMODEL / NHEAD
#define BATCH   16
#define NSEQ    8192
#define TILE_N  64                 // rows of x per workgroup
#define CHUNKS  (NSEQ / TILE_N)    // 128 chunks per batch
#define RT      (TILE_N / 16)      // 4 row tiles per wave
#define PSTRIDE 66                 // per-(b,chunk,h) partial: m, l, acc[64]
#define WELEMS  (DMODEL * DMODEL)  // 262144 elements per W matrix

typedef float  v8f   __attribute__((ext_vector_type(8)));
typedef __bf16 v4bf  __attribute__((ext_vector_type(4)));
typedef __bf16 v8bf  __attribute__((ext_vector_type(8)));
typedef __bf16 v16bf __attribute__((ext_vector_type(16)));

// D(16x16 f32) += A(16x32 bf16) * B(32x16 bf16)   [codegen-confirmed builtin]
__device__ __forceinline__ v8f wmma_bf16(v16bf a, v16bf b, v8f c) {
    return __builtin_amdgcn_wmma_f32_16x16x32_bf16(
        /*neg_a=*/false, a, /*neg_b=*/false, b,
        /*c_mod=*/(short)0, c, /*reuse_a=*/false, /*reuse_b=*/false);
}

// Per-lane 16-bit WMMA operand: K = {k0+8h .. +7} ++ {k0+16+8h .. +7}
// (p already points at base + k0 + 8h; both halves are contiguous b128 loads)
__device__ __forceinline__ v16bf load_op(const __bf16* p) {
    const v8bf lo = *(const v8bf*)(p);
    const v8bf hi = *(const v8bf*)(p + 16);
    return __builtin_shufflevector(lo, hi, 0,1,2,3,4,5,6,7,8,9,10,11,12,13,14,15);
}

// ---------------------------------------------------------------------------
// Phase 0: convert W_k / W_g fp32 -> bf16 into workspace (run once, tiny)
// ---------------------------------------------------------------------------
__global__ __launch_bounds__(256)
void convert_w(const float* __restrict__ Wk, const float* __restrict__ Wg,
               __bf16* __restrict__ wkbf, __bf16* __restrict__ wgbf)
{
    const int i = (blockIdx.x * 256 + threadIdx.x) * 4;
    const float4 a = *(const float4*)(Wk + i);
    const float4 b = *(const float4*)(Wg + i);
    const v4bf ak = { (__bf16)a.x, (__bf16)a.y, (__bf16)a.z, (__bf16)a.w };
    const v4bf bg = { (__bf16)b.x, (__bf16)b.y, (__bf16)b.z, (__bf16)b.w };
    *(v4bf*)(wkbf + i) = ak;
    *(v4bf*)(wgbf + i) = bg;
}

// ---------------------------------------------------------------------------
// Phase 1: per-chunk fused  K/G bf16-GEMM -> gate -> score -> partial pool
// grid = BATCH*CHUNKS blocks of 256 threads (8 waves). Wave w == head w.
// LDS: x fp32 tile (pooling) + x bf16 tile (WMMA A) + scores
// ---------------------------------------------------------------------------
__global__ __launch_bounds__(256)
void gated_pool_phase1(const float* __restrict__ x,
                       const __bf16* __restrict__ wkbf,
                       const __bf16* __restrict__ wgbf,
                       const float* __restrict__ q,
                       float* __restrict__ partials)
{
    extern __shared__ float smem[];
    float*  xs = smem;                                  // TILE_N*DMODEL fp32
    __bf16* xb = (__bf16*)(smem + TILE_N * DMODEL);     // TILE_N*DMODEL bf16
    float*  sc = (float*)(xb + TILE_N * DMODEL);        // TILE_N*NHEAD

    const int blk   = blockIdx.x;
    const int b     = blk / CHUNKS;
    const int chunk = blk % CHUNKS;
    const int n0    = chunk * TILE_N;
    const int tid   = threadIdx.x;

    // ---- Stage 0: stage x tile into LDS (fp32 + bf16 copies), zero scores
    const float* xg = x + ((size_t)b * NSEQ + n0) * DMODEL;
    for (int i = tid * 8; i < TILE_N * DMODEL; i += 256 * 8) {
        const float4 a = *(const float4*)(xg + i);
        const float4 c = *(const float4*)(xg + i + 4);
        *(float4*)(xs + i)     = a;
        *(float4*)(xs + i + 4) = c;
        const v8bf o = { (__bf16)a.x, (__bf16)a.y, (__bf16)a.z, (__bf16)a.w,
                         (__bf16)c.x, (__bf16)c.y, (__bf16)c.z, (__bf16)c.w };
        *(v8bf*)(xb + i) = o;
    }
    for (int i = tid; i < TILE_N * NHEAD; i += 256) sc[i] = 0.0f;
    __syncthreads();

    const int lane = tid & 31;
    const int wave = tid >> 5;        // == head index for this wave
    const int hv   = lane >> 4;       // lane half
    const int lr   = lane & 15;

    // ---- Stage 1: K = x@Wk^T, G = x@Wg^T for this wave's 4 column tiles
    for (int ci = 0; ci < 4; ++ci) {
        const int c    = wave * 4 + ci;   // column tile; head = c/4 = wave
        const int col0 = c * 16;

        v8f accK[RT], accG[RT];
        #pragma unroll
        for (int rt = 0; rt < RT; ++rt) { v8f z = {}; accK[rt] = z; accG[rt] = z; }

        const __bf16* wkp = wkbf + (size_t)(col0 + lr) * DMODEL + 8 * hv;
        const __bf16* wgp = wgbf + (size_t)(col0 + lr) * DMODEL + 8 * hv;
        const __bf16* xap = xb + lr * DMODEL + 8 * hv;

        // software pipeline: operands for step k live in a*/b*, step k+32
        // prefetched into p* while the 8 WMMAs of step k execute.
        v16bf a0 = load_op(xap);
        v16bf a1 = load_op(xap + 16 * DMODEL);
        v16bf a2 = load_op(xap + 32 * DMODEL);
        v16bf a3 = load_op(xap + 48 * DMODEL);
        v16bf bk = load_op(wkp);
        v16bf bg = load_op(wgp);

        #pragma unroll 2
        for (int k = 0; k < DMODEL; k += 32) {
            // Final-iteration over-read is benign: LDS OOB loads return 0 and
            // the W over-read stays inside the workspace; values never used.
            const int kn = k + 32;
            const v16bf pa0 = load_op(xap + kn);
            const v16bf pa1 = load_op(xap + 16 * DMODEL + kn);
            const v16bf pa2 = load_op(xap + 32 * DMODEL + kn);
            const v16bf pa3 = load_op(xap + 48 * DMODEL + kn);
            const v16bf pbk = load_op(wkp + kn);
            const v16bf pbg = load_op(wgp + kn);

            accK[0] = wmma_bf16(a0, bk, accK[0]);
            accG[0] = wmma_bf16(a0, bg, accG[0]);
            accK[1] = wmma_bf16(a1, bk, accK[1]);
            accG[1] = wmma_bf16(a1, bg, accG[1]);
            accK[2] = wmma_bf16(a2, bk, accK[2]);
            accG[2] = wmma_bf16(a2, bg, accG[2]);
            accK[3] = wmma_bf16(a3, bk, accK[3]);
            accG[3] = wmma_bf16(a3, bg, accG[3]);

            a0 = pa0; a1 = pa1; a2 = pa2; a3 = pa3; bk = pbk; bg = pbg;
        }

        // gate + query dot; C layout: vgpr vv, lane -> row = 8*hv+vv (+rt*16), col = lr
        const float qv = q[wave * DKH + ci * 16 + lr];
        #pragma unroll
        for (int rt = 0; rt < RT; ++rt) {
            #pragma unroll
            for (int vv = 0; vv < 8; ++vv) {
                const float kk = accK[rt][vv];
                const float gg = accG[rt][vv];
                // branch-free: tanh(k) = 1 - 2/(1+e^{2k}), sigmoid via rcp
                const float th = 1.0f - 2.0f * __builtin_amdgcn_rcpf(1.0f + __expf(2.0f * kk));
                const float sg = __builtin_amdgcn_rcpf(1.0f + __expf(-gg));
                float s = th * sg * qv;
                // reduce across the 16 lanes of this half (columns of the tile)
                s += __shfl_xor(s, 1);
                s += __shfl_xor(s, 2);
                s += __shfl_xor(s, 4);
                s += __shfl_xor(s, 8);
                if (lr == 0) {
                    sc[(rt * 16 + 8 * hv + vv) * NHEAD + wave] += s;
                }
            }
        }
    }
    __syncthreads();

    // ---- Stage 2: per-head partial softmax + weighted pooling (fp32 x tile)
    const int h = wave;
    float m = -1e30f;
    #pragma unroll
    for (int r = 0; r < TILE_N / 32; ++r) m = fmaxf(m, sc[(lane + 32 * r) * NHEAD + h]);
    #pragma unroll
    for (int off = 16; off >= 1; off >>= 1) m = fmaxf(m, __shfl_xor(m, off));

    float l = 0.0f;
    #pragma unroll
    for (int r = 0; r < TILE_N / 32; ++r) l += __expf(sc[(lane + 32 * r) * NHEAD + h] - m);
    #pragma unroll
    for (int off = 16; off >= 1; off >>= 1) l += __shfl_xor(l, off);

    const int dk0 = lane * 2;                 // 2 dk per lane -> 64 per wave
    float2 acc = make_float2(0.0f, 0.0f);
    for (int r = 0; r < TILE_N; ++r) {
        const float wgt = __expf(sc[r * NHEAD + h] - m);
        const float2 xv = *(const float2*)(xs + r * DMODEL + h * DKH + dk0);
        acc.x += wgt * xv.x;
        acc.y += wgt * xv.y;
    }

    float* p = partials + (size_t)((b * CHUNKS + chunk) * NHEAD + h) * PSTRIDE;
    if (lane == 0) { p[0] = m; p[1] = l; }
    *(float2*)(p + 2 + dk0) = acc;
}

// ---------------------------------------------------------------------------
// Phase 2: log-sum-exp merge of chunk partials. grid = BATCH*NHEAD waves.
// ---------------------------------------------------------------------------
__global__ __launch_bounds__(32)
void gated_pool_phase2(const float* __restrict__ partials,
                       float* __restrict__ out)
{
    const int bh = blockIdx.x;              // 0 .. BATCH*NHEAD-1
    const int b = bh / NHEAD, h = bh % NHEAD;
    const int lane = threadIdx.x;
    const float* pb = partials + (size_t)(b * CHUNKS * NHEAD + h) * PSTRIDE;
    const size_t cs = (size_t)NHEAD * PSTRIDE;   // stride between chunks

    float M = -1e30f;
    for (int c = 0; c < CHUNKS; ++c) M = fmaxf(M, pb[c * cs]);

    const int dk0 = lane * 2;
    float L = 0.0f;
    float2 acc = make_float2(0.0f, 0.0f);
    for (int c = 0; c < CHUNKS; ++c) {
        const float* p = pb + c * cs;
        const float w = __expf(p[0] - M);
        L += p[1] * w;
        const float2 a = *(const float2*)(p + 2 + dk0);
        acc.x += w * a.x;
        acc.y += w * a.y;
    }
    const float inv = 1.0f / L;
    out[(size_t)b * DMODEL + h * DKH + dk0]     = acc.x * inv;
    out[(size_t)b * DMODEL + h * DKH + dk0 + 1] = acc.y * inv;
}

// ---------------------------------------------------------------------------
extern "C" void kernel_launch(void* const* d_in, const int* in_sizes, int n_in,
                              void* d_out, int out_size, void* d_ws, size_t ws_size,
                              hipStream_t stream) {
    const float* x  = (const float*)d_in[0];   // (B, N, D)
    const float* Wk = (const float*)d_in[1];   // (D, D)
    const float* Wg = (const float*)d_in[2];   // (D, D)
    const float* q  = (const float*)d_in[3];   // (H, 1, DK)
    float* out      = (float*)d_out;           // (B, D)

    // workspace layout: [Wk bf16][Wg bf16][partials f32]
    __bf16* wkbf = (__bf16*)d_ws;
    __bf16* wgbf = wkbf + WELEMS;
    float* partials = (float*)(wgbf + WELEMS);  // B*CHUNKS*H*PSTRIDE floats (~4.3 MB)

    convert_w<<<WELEMS / 4 / 256, 256, 0, stream>>>(Wk, Wg, wkbf, wgbf);

    const size_t shmem = (size_t)TILE_N * DMODEL * sizeof(float)      // fp32 tile
                       + (size_t)TILE_N * DMODEL * sizeof(__bf16)     // bf16 tile
                       + (size_t)TILE_N * NHEAD * sizeof(float);      // scores
    gated_pool_phase1<<<BATCH * CHUNKS, 256, shmem, stream>>>(x, wkbf, wgbf, q, partials);
    gated_pool_phase2<<<BATCH * NHEAD, 32, 0, stream>>>(partials, out);
}
